// VectorQuantizer_26439818674910
// MI455X (gfx1250) — matrix operand order repaired and checked by reading, not compiled, not run on hardware
//
#include <hip/hip_runtime.h>
#include <hip/hip_bf16.h>

// VQ: x[16,1024,256] f32, codebook[8192,256] f32.
// outputs concat: x_q_st (4194304 f32), idx (16384 as f32), vq_loss (1 f32).

typedef __attribute__((ext_vector_type(16))) __bf16 v16bf;
typedef __attribute__((ext_vector_type(8)))  __bf16 v8bf;
typedef __attribute__((ext_vector_type(8)))  float  v8f;

#define N_ROWS 16384
#define DDIM   256
#define KCB    8192
#define CHUNK  32          // codebook rows staged per iteration (16 KB bf16)

// ---------------- prep: f32 -> bf16 (RNE) + e2 ----------------
__device__ __forceinline__ unsigned short f2bf(float f) {
  unsigned u = __float_as_uint(f);
  u += 0x7FFFu + ((u >> 16) & 1u);
  return (unsigned short)(u >> 16);
}

__global__ __launch_bounds__(256)
void vq_prep_kernel(const float* __restrict__ x, const float* __restrict__ cb,
                    unsigned short* __restrict__ xbf, unsigned short* __restrict__ cbbf,
                    float* __restrict__ e2) {
  const int i = blockIdx.x * 256 + threadIdx.x;   // grid covers N_ROWS*DDIM
  xbf[i] = f2bf(x[i]);
  if (i < KCB * DDIM) cbbf[i] = f2bf(cb[i]);
  if (i < KCB) {
    float s = 0.f;
    const float* r = cb + (size_t)i * DDIM;
    #pragma unroll 8
    for (int d = 0; d < DDIM; ++d) s += r[d] * r[d];
    e2[i] = s;
  }
}

// ---------------- main: async-LDS staged WMMA distance + fused argmin ----------------
__global__ __launch_bounds__(256)
void vq_argmin_kernel(const __bf16* __restrict__ xbf,
                      const __bf16* __restrict__ cbbf,
                      const float* __restrict__ e2,
                      int* __restrict__ idx_out) {
  __shared__ float  s_e2[KCB];                  // 32 KB
  __shared__ __bf16 s_cb[2][CHUNK * DDIM];      // 2 x 16 KB double buffer

  const int t    = threadIdx.x;
  for (int i = t; i < KCB; i += 256) s_e2[i] = e2[i];

  const int lane = t & 31;
  const int wv   = t >> 5;                      // 8 waves / block
  const int hb   = lane >> 4;                   // half-wave 0/1
  const int lg   = lane & 15;
  const int row0 = (blockIdx.x * 8 + wv) * 16;  // 16-row tile per wave
  const int arow = row0 + lg;

  // A fragments (16x32 bf16 per chunk), contiguous layout:
  //   lanes 0-15 row M=lane: K 0-15 in elems 0-15
  //   lanes 16-31 row M=lane-16: K 16-31
  union AF { v16bf v; v8bf h[2]; };
  AF afrag[8];
  const __bf16* xrow = xbf + (size_t)arow * DDIM + hb * 16;
  #pragma unroll
  for (int c = 0; c < 8; ++c) {
    afrag[c].h[0] = *(const v8bf*)(xrow + c * 32);
    afrag[c].h[1] = *(const v8bf*)(xrow + c * 32 + 8);
  }

  // per-thread staging slots: 64 B each (4 x b128), tile is contiguous 16 KB in global
  const unsigned lds0 = (unsigned)(size_t)(void*)&s_cb[0][0] + (unsigned)t * 64u;
  const unsigned lds1 = (unsigned)(size_t)(void*)&s_cb[1][0] + (unsigned)t * 64u;

  // prefetch tile 0 into buffer 0 (ASYNCcnt-tracked)
  {
    const char* g = (const char*)cbbf + (size_t)t * 64;
    #pragma unroll
    for (int j = 0; j < 4; ++j)
      asm volatile("global_load_async_to_lds_b128 %0, %1, off"
                   :: "v"(lds0 + j * 16), "v"(g + j * 16) : "memory");
  }

  float best[8];
  int   bidx[8];
  #pragma unroll
  for (int v = 0; v < 8; ++v) { best[v] = 3.4e38f; bidx[v] = 0; }

  for (int k0 = 0; k0 < KCB; k0 += CHUNK) {
    const int buf = (k0 >> 5) & 1;
    __syncthreads();                            // buf^1 free (all waves past its compute)

    // prefetch next tile into buf^1 (dummy-wrap on last iteration keeps counters uniform)
    const int knext = (k0 + CHUNK < KCB) ? (k0 + CHUNK) : 0;
    const unsigned ldsn = buf ? lds0 : lds1;
    const char* g = (const char*)(cbbf + (size_t)knext * DDIM) + (size_t)t * 64;
    #pragma unroll
    for (int j = 0; j < 4; ++j)
      asm volatile("global_load_async_to_lds_b128 %0, %1, off"
                   :: "v"(ldsn + j * 16), "v"(g + j * 16) : "memory");

    asm volatile("s_wait_asynccnt 0x4" ::: "memory");  // my tile-k0 stores to LDS retired
    __syncthreads();                                   // everyone's retired -> tile resident

    const __bf16* bb = &s_cb[buf][0];
    #pragma unroll
    for (int g2 = 0; g2 < 2; ++g2) {            // two 16-column groups per staged tile
      // B (32x16 bf16): lane = one column (codebook row), halves hold K 0-15 / 16-31
      const __bf16* bptr = bb + (g2 * 16 + lg) * DDIM + hb * 16;
      v8f acc = {};
      #pragma unroll
      for (int c = 0; c < 8; ++c) {             // accumulate over D=256 from LDS
        v16bf b = *(const v16bf*)(bptr + c * 32);
        acc = __builtin_amdgcn_wmma_f32_16x16x32_bf16(
            false, afrag[c].v, false, b, (short)0, acc, false, false);
      }
      // C layout: lane L, vgpr v -> (row M = v + 8*hb, col N = lg). dist = e2 - 2*x.e
      const int krow = k0 + g2 * 16 + lg;
      const float e2v = s_e2[krow];
      #pragma unroll
      for (int v = 0; v < 8; ++v) {
        float d = e2v - 2.0f * acc[v];
        if (d < best[v]) { best[v] = d; bidx[v] = krow; }
      }
    }
  }

  // argmin across the 16 column-lanes (stay inside each 16-lane group: xor 1,2,4,8)
  #pragma unroll
  for (int v = 0; v < 8; ++v) {
    #pragma unroll
    for (int m = 1; m < 16; m <<= 1) {
      float od = __shfl_xor(best[v], m, 32);
      int   oi = __shfl_xor(bidx[v], m, 32);
      if (od < best[v] || (od == best[v] && oi < bidx[v])) { best[v] = od; bidx[v] = oi; }
    }
  }
  if (lg == 0) {                                // lanes 0 (rows 0-7) and 16 (rows 8-15)
    #pragma unroll
    for (int v = 0; v < 8; ++v) idx_out[row0 + hb * 8 + v] = bidx[v];
  }
}

// ---------------- gather x_q (f32), idx-as-float, per-row loss ----------------
__global__ __launch_bounds__(256)
void vq_gather_kernel(const float* __restrict__ x, const float* __restrict__ cb,
                      const int* __restrict__ idx, float* __restrict__ out,
                      float* __restrict__ rowloss) {
  const int n = blockIdx.x;
  const int t = threadIdx.x;                    // t = d (DDIM == 256 == blockDim)
  const int k = idx[n];
  const float q  = cb[(size_t)k * DDIM + t];
  const float xv = x[(size_t)n * DDIM + t];
  out[(size_t)n * DDIM + t] = q;
  float s = (q - xv) * (q - xv);
  #pragma unroll
  for (int m = 1; m < 32; m <<= 1) s += __shfl_xor(s, m, 32);
  __shared__ float red[8];
  if ((t & 31) == 0) red[t >> 5] = s;
  __syncthreads();
  if (t == 0) {
    float tot = 0.f;
    #pragma unroll
    for (int w = 0; w < 8; ++w) tot += red[w];
    rowloss[n] = tot;
    out[(size_t)N_ROWS * DDIM + n] = (float)k;  // idx output slot
  }
}

// ---------------- deterministic final loss reduction ----------------
__global__ __launch_bounds__(256)
void vq_loss_kernel(const float* __restrict__ rowloss, float* __restrict__ out) {
  __shared__ float sm[256];
  float s = 0.f;
  for (int i = threadIdx.x; i < N_ROWS; i += 256) s += rowloss[i];
  sm[threadIdx.x] = s;
  __syncthreads();
  for (int step = 128; step > 0; step >>= 1) {
    if (threadIdx.x < step) sm[threadIdx.x] += sm[threadIdx.x + step];
    __syncthreads();
  }
  if (threadIdx.x == 0) {
    // vq_loss = mse + BETA*mse = 1.25 * sum / (N*D)
    out[(size_t)N_ROWS * DDIM + N_ROWS] = sm[0] * (1.25f / (float)(N_ROWS * DDIM));
  }
}

extern "C" void kernel_launch(void* const* d_in, const int* in_sizes, int n_in,
                              void* d_out, int out_size, void* d_ws, size_t ws_size,
                              hipStream_t stream) {
  const float* x  = (const float*)d_in[0];   // [16,1024,256]
  const float* cb = (const float*)d_in[1];   // [8192,256]
  float* out = (float*)d_out;

  // workspace layout (bytes)
  char* ws = (char*)d_ws;
  unsigned short* xbf    = (unsigned short*)(ws);                         // 8,388,608 B
  unsigned short* cbbf   = (unsigned short*)(ws + 8388608);               // 4,194,304 B
  float*          e2     = (float*)(ws + 12582912);                       //    32,768 B
  int*            idxw   = (int*)(ws + 12615680);                         //    65,536 B
  float*          rloss  = (float*)(ws + 12681216);                       //    65,536 B

  vq_prep_kernel<<<(N_ROWS * DDIM) / 256, 256, 0, stream>>>(x, cb, xbf, cbbf, e2);
  vq_argmin_kernel<<<N_ROWS / (16 * 8), 256, 0, stream>>>(
      (const __bf16*)xbf, (const __bf16*)cbbf, e2, idxw);
  vq_gather_kernel<<<N_ROWS, 256, 0, stream>>>(x, cb, idxw, out, rloss);
  vq_loss_kernel<<<1, 256, 0, stream>>>(rloss, out);
}
